// ParticleMatcher_55362128445847
// MI455X (gfx1250) — matrix operand order repaired and checked by reading, compiled, not run on hardware
//
#include <hip/hip_runtime.h>

typedef _Float16 half_t;
typedef __attribute__((ext_vector_type(16))) _Float16 v16h;
typedef __attribute__((ext_vector_type(8)))  _Float16 v8h;
typedef __attribute__((ext_vector_type(8)))  float    v8f;

// A fragment: 16 halves from W row: K in [0,8) at p, K in [16,24) at p+16
__device__ __forceinline__ v16h load_frag_a(const half_t* __restrict__ p) {
  v16h a;
  const v8h lo = *(const v8h*)(p);
  const v8h hi8 = *(const v8h*)(p + 16);
#pragma unroll
  for (int i = 0; i < 8; ++i) { a[i] = lo[i]; a[i + 8] = hi8[i]; }
  return a;
}

// B fragment: 16 halves from X row: K in [0,8) at p, K in [16,24) at p+16
__device__ __forceinline__ v16h load_frag_b(const half_t* __restrict__ p) {
  v16h b;
  const v8h b0 = *(const v8h*)(p);
  const v8h b1 = *(const v8h*)(p + 16);
#pragma unroll
  for (int i = 0; i < 8; ++i) { b[i] = b0[i]; b[i + 8] = b1[i]; }
  return b;
}

// ---------------------------------------------------------------------------
// Generic WMMA GEMM:
//   C[o][n] = alpha * sum_k W[g][o][k] * X[g][n][k]  (+ bias[o]) (+ residual)
// W: f16 [G][O][K], row stride ldw, group stride wgs
// X: f16 [G][N][K], row stride ldx, group stride xgs
// store address = ooff + (g/gdiv)*ogsA + (g%gdiv)*ogsB + o*ostr + n*nstr
// exactly one of outH (f16) / outF (f32) is non-null.
// One wave per block computes a 16(o) x 64(n) tile; software pipelined with
// a peeled final WMMA group (K is always a multiple of 64 here, so the loop
// body executes at least once).
// ---------------------------------------------------------------------------
__global__ __launch_bounds__(32)
void wmma_gemm_kernel(const half_t* __restrict__ W, long long wgs, int ldw,
                      const half_t* __restrict__ X, long long xgs, int ldx,
                      const float*  __restrict__ bias,
                      half_t* __restrict__ outH, float* __restrict__ outF,
                      int gdiv, long long ogsA, long long ogsB,
                      long long ostr, long long nstr, long long ooff,
                      int K, float alpha, int residual)
{
  const int lane  = threadIdx.x;
  const int nblk  = blockIdx.x;   // 64-wide column block
  const int otile = blockIdx.y;   // 16-wide row block
  const int g     = blockIdx.z;

  const half_t* Wg = W + (long long)g * wgs;
  const half_t* Xg = X + (long long)g * xgs;
  const long long obase = ooff + (long long)(g / gdiv) * ogsA
                               + (long long)(g % gdiv) * ogsB;

  const int row = lane & 15;   // A: row M within tile; B/C: column within subtile
  const int hi  = lane >> 4;   // half-wave select

  const half_t* Arow = Wg + (long long)(otile * 16 + row) * ldw + hi * 8;
  const half_t* B0 = Xg + (long long)(nblk * 64 +  0 + row) * ldx + hi * 8;
  const half_t* B1 = Xg + (long long)(nblk * 64 + 16 + row) * ldx + hi * 8;
  const half_t* B2 = Xg + (long long)(nblk * 64 + 32 + row) * ldx + hi * 8;
  const half_t* B3 = Xg + (long long)(nblk * 64 + 48 + row) * ldx + hi * 8;

  // Prologue: fragments for kk = 0
  v16h a  = load_frag_a(Arow);
  v16h b0 = load_frag_b(B0);
  v16h b1 = load_frag_b(B1);
  v16h b2 = load_frag_b(B2);
  v16h b3 = load_frag_b(B3);

  v8f acc[4] = {};

  // Pipelined main loop: issue loads for slab kk, then consume slab kk-32.
  for (int kk = 32; kk < K; kk += 32) {
    if (kk + 32 < K) __builtin_prefetch(Arow + kk + 32, 0, 1);
    v16h an  = load_frag_a(Arow + kk);
    v16h bn0 = load_frag_b(B0 + kk);
    v16h bn1 = load_frag_b(B1 + kk);
    v16h bn2 = load_frag_b(B2 + kk);
    v16h bn3 = load_frag_b(B3 + kk);

    acc[0] = __builtin_amdgcn_wmma_f32_16x16x32_f16(false, a, false, b0,
                                                    (short)0, acc[0], false, false);
    acc[1] = __builtin_amdgcn_wmma_f32_16x16x32_f16(false, a, false, b1,
                                                    (short)0, acc[1], false, false);
    acc[2] = __builtin_amdgcn_wmma_f32_16x16x32_f16(false, a, false, b2,
                                                    (short)0, acc[2], false, false);
    acc[3] = __builtin_amdgcn_wmma_f32_16x16x32_f16(false, a, false, b3,
                                                    (short)0, acc[3], false, false);
    a = an; b0 = bn0; b1 = bn1; b2 = bn2; b3 = bn3;
  }
  // Peeled final WMMA group (no guarded loads, no clamp logic).
  acc[0] = __builtin_amdgcn_wmma_f32_16x16x32_f16(false, a, false, b0,
                                                  (short)0, acc[0], false, false);
  acc[1] = __builtin_amdgcn_wmma_f32_16x16x32_f16(false, a, false, b1,
                                                  (short)0, acc[1], false, false);
  acc[2] = __builtin_amdgcn_wmma_f32_16x16x32_f16(false, a, false, b2,
                                                  (short)0, acc[2], false, false);
  acc[3] = __builtin_amdgcn_wmma_f32_16x16x32_f16(false, a, false, b3,
                                                  (short)0, acc[3], false, false);

  // Epilogue. C/D layout: lane -> col = lane&15, rows = (lane>>4)*8 + r.
  const long long orow0 = otile * 16 + hi * 8;
  float bv[8];
#pragma unroll
  for (int r = 0; r < 8; ++r) bv[r] = bias ? bias[orow0 + r] : 0.0f;

#pragma unroll
  for (int s = 0; s < 4; ++s) {
    const long long ncol = nblk * 64 + s * 16 + row;
    if (outF) {
#pragma unroll
      for (int r = 0; r < 8; ++r) {
        outF[obase + (orow0 + r) * ostr + ncol * nstr] = acc[s][r] * alpha + bv[r];
      }
    } else if (ostr == 1) {
      // 8 consecutive halves per lane -> one b128 store (aligned by layout)
      half_t* p = outH + obase + orow0 + ncol * nstr;
      v8h val;
      if (residual) {
        const v8h old = *(const v8h*)p;
#pragma unroll
        for (int r = 0; r < 8; ++r) {
          val[r] = (half_t)(acc[s][r] * alpha + bv[r] + (float)old[r]);
        }
      } else {
#pragma unroll
        for (int r = 0; r < 8; ++r) {
          val[r] = (half_t)(acc[s][r] * alpha + bv[r]);
        }
      }
      *(v8h*)p = val;
    } else {
#pragma unroll
      for (int r = 0; r < 8; ++r) {
        float c = acc[s][r] * alpha + bv[r];
        const long long addr = obase + (orow0 + r) * ostr + ncol * nstr;
        if (residual) c += (float)outH[addr];
        outH[addr] = (half_t)c;
      }
    }
  }
}

// ---------------------------------------------------------------------------
// Elementwise helpers
// ---------------------------------------------------------------------------
__global__ void cvt_f32_to_f16(const float* __restrict__ s,
                               half_t* __restrict__ d, long long n) {
  long long i = (long long)blockIdx.x * blockDim.x + threadIdx.x;
  if (i < n) d[i] = (half_t)s[i];
}

// x [B=4, D=256, N=512] f32 -> cat[b][n][0:256] f16 (row stride 512)
__global__ void transpose_in(const float* __restrict__ x,
                             half_t* __restrict__ cat) {
  long long i = (long long)blockIdx.x * blockDim.x + threadIdx.x;
  if (i >= 4LL * 256 * 512) return;
  int n = (int)(i & 511);
  int d = (int)((i >> 9) & 255);
  int b = (int)(i >> 17);
  cat[((long long)(b * 512 + n)) * 512 + d] = (half_t)x[i];
}

// qT/kT/vT [b][n][c] (c = dd*4 + h) -> head-major layouts
__global__ void repack_qkv(const half_t* __restrict__ qT,
                           const half_t* __restrict__ kT,
                           const half_t* __restrict__ vT,
                           half_t* __restrict__ qh,
                           half_t* __restrict__ kh,
                           half_t* __restrict__ vhT) {
  long long i = (long long)blockIdx.x * blockDim.x + threadIdx.x;
  if (i >= 4LL * 512 * 256) return;
  int c = (int)(i & 255);
  int n = (int)((i >> 8) & 511);
  int b = (int)(i >> 17);
  int h = c & 3, dd = c >> 2;
  long long dq = (((long long)(b * 4 + h) * 512) + n) * 64 + dd;  // [g][n][dd]
  long long dv = (((long long)(b * 4 + h) * 64) + dd) * 512 + n;  // [g][dd][n]
  qh[dq] = qT[i];
  kh[dq] = kT[i];
  vhT[dv] = vT[i];
}

// softmax over rows of S [8192][512] f32 -> P f16, one wave per row
__global__ __launch_bounds__(128)
void softmax_rows(const float* __restrict__ S, half_t* __restrict__ P) {
  int wave = (int)((blockIdx.x * (long long)blockDim.x + threadIdx.x) >> 5);
  int lane = threadIdx.x & 31;
  if (wave >= 16 * 512) return;
  const float* row = S + (long long)wave * 512;
  float m = -1e30f;
  for (int j = lane; j < 512; j += 32) m = fmaxf(m, row[j]);
  for (int o = 16; o; o >>= 1) m = fmaxf(m, __shfl_xor(m, o, 32));
  float sum = 0.0f;
  for (int j = lane; j < 512; j += 32) sum += __expf(row[j] - m);
  for (int o = 16; o; o >>= 1) sum += __shfl_xor(sum, o, 32);
  float inv = 1.0f / sum;
  half_t* prow = P + (long long)wave * 512;
  for (int j = lane; j < 512; j += 32)
    prow[j] = (half_t)(__expf(row[j] - m) * inv);
}

// h = relu(h * g[c]*scale + b[c]) in place, h [4*512][512] f16
__global__ void bnrelu(half_t* __restrict__ h, const float* __restrict__ g,
                       const float* __restrict__ b, float scale) {
  long long i = (long long)blockIdx.x * blockDim.x + threadIdx.x;
  if (i >= 4LL * 512 * 512) return;
  int c = (int)(i & 511);
  float x = (float)h[i] * (g[c] * scale) + b[c];
  h[i] = (half_t)fmaxf(x, 0.0f);
}

// fill dustbin row/col of Z0 [4][513][513]
__global__ void fill_bins(float* __restrict__ Z0, const float* __restrict__ bin) {
  int i = blockIdx.x * blockDim.x + threadIdx.x;
  if (i >= 4 * 513) return;
  int b = i / 513, k = i % 513;
  float val = *bin;
  float* Zb = Z0 + (long long)b * 513 * 513;
  Zb[512LL * 513 + k] = val;
  Zb[(long long)k * 513 + 512] = val;
}

__global__ void zero_f32(float* __restrict__ p, long long n) {
  long long i = (long long)blockIdx.x * blockDim.x + threadIdx.x;
  if (i < n) p[i] = 0.0f;
}

// u[b][i] = log_mu[i] - LSE_j(Z0[b][i][j] + v[b][j]); one wave per row
__global__ __launch_bounds__(128)
void sink_u(const float* __restrict__ Z0, const float* __restrict__ v,
            float* __restrict__ u) {
  int wave = (int)((blockIdx.x * (long long)blockDim.x + threadIdx.x) >> 5);
  int lane = threadIdx.x & 31;
  if (wave >= 4 * 513) return;
  int b = wave / 513, i = wave % 513;
  const float* row = Z0 + ((long long)b * 513 + i) * 513;
  const float* vb = v + b * 513;
  float m = -1e30f, s = 0.0f;
  for (int j = lane; j < 513; j += 32) {
    float z = row[j] + vb[j];
    if (z > m) { s = s * __expf(m - z) + 1.0f; m = z; }
    else       { s += __expf(z - m); }
  }
  for (int o = 16; o; o >>= 1) {
    float m2 = __shfl_xor(m, o, 32), s2 = __shfl_xor(s, o, 32);
    float M = fmaxf(m, m2);
    s = s * __expf(m - M) + s2 * __expf(m2 - M);
    m = M;
  }
  if (lane == 0) {
    const float norm = -logf(1024.0f);
    float logmu = (i == 512) ? (logf(512.0f) + norm) : norm;
    u[b * 513 + i] = logmu - (logf(s) + m);
  }
}

// v[b][j] = log_nu[j] - LSE_i(Z0[b][i][j] + u[b][i]); thread per column
__global__ __launch_bounds__(256)
void sink_v(const float* __restrict__ Z0, const float* __restrict__ u,
            float* __restrict__ v) {
  int j = blockIdx.x * 256 + threadIdx.x;
  int b = blockIdx.y;
  if (j >= 513) return;
  const float* Zb = Z0 + (long long)b * 513 * 513;
  const float* ub = u + b * 513;
  float m = -1e30f, s = 0.0f;
  for (int i = 0; i < 513; ++i) {
    float z = Zb[(long long)i * 513 + j] + ub[i];
    if (z > m) { s = s * __expf(m - z) + 1.0f; m = z; }
    else       { s += __expf(z - m); }
  }
  const float norm = -logf(1024.0f);
  float lognu = (j == 512) ? (logf(512.0f) + norm) : norm;
  v[b * 513 + j] = lognu - (logf(s) + m);
}

__global__ void sink_final(const float* __restrict__ Z0,
                           const float* __restrict__ u,
                           const float* __restrict__ v,
                           float* __restrict__ out) {
  long long idx = (long long)blockIdx.x * blockDim.x + threadIdx.x;
  if (idx >= 4LL * 513 * 513) return;
  int j = (int)(idx % 513);
  long long t = idx / 513;
  int i = (int)(t % 513);
  int b = (int)(t / 513);
  const float norm = -logf(1024.0f);
  out[idx] = __expf(Z0[idx] + u[b * 513 + i] + v[b * 513 + j] - norm);
}

// ---------------------------------------------------------------------------
// Host orchestration
// ---------------------------------------------------------------------------
extern "C" void kernel_launch(void* const* d_in, const int* in_sizes, int n_in,
                              void* d_out, int out_size, void* d_ws,
                              size_t ws_size, hipStream_t stream) {
  (void)in_sizes; (void)n_in; (void)out_size; (void)ws_size;

  const float* x0_in    = (const float*)d_in[0];
  const float* x1_in    = (const float*)d_in[1];
  const float* proj_w   = (const float*)d_in[2];
  const float* proj_b   = (const float*)d_in[3];
  const float* merge_w  = (const float*)d_in[4];
  const float* merge_b  = (const float*)d_in[5];
  const float* mlp1_w   = (const float*)d_in[6];
  const float* mlp1_b   = (const float*)d_in[7];
  const float* bn_g     = (const float*)d_in[8];
  const float* bn_b     = (const float*)d_in[9];
  const float* mlp2_w   = (const float*)d_in[10];
  const float* mlp2_b   = (const float*)d_in[11];
  const float* final_w  = (const float*)d_in[12];
  const float* final_b  = (const float*)d_in[13];
  const float* binscore = (const float*)d_in[14];

  const int Bsz = 4, Dm = 256, Nn = 512, Ll = 18;
  const float bn_scale = 1.0f / sqrtf(1.0f + 1e-5f);

  // workspace carve-out
  char* base = (char*)d_ws;
  size_t off = 0;
  auto alloc = [&](size_t bytes) -> void* {
    void* p = base + off;
    off = (off + bytes + 255) & ~(size_t)255;
    return p;
  };
  half_t* projW  = (half_t*)alloc((size_t)Ll * 3 * Dm * Dm * 2);
  half_t* mergeW = (half_t*)alloc((size_t)Ll * Dm * Dm * 2);
  half_t* mlp1W  = (half_t*)alloc((size_t)Ll * 2 * Dm * 2 * Dm * 2);
  half_t* mlp2W  = (half_t*)alloc((size_t)Ll * Dm * 2 * Dm * 2);
  half_t* finW   = (half_t*)alloc((size_t)Dm * Dm * 2);
  half_t* cat0   = (half_t*)alloc((size_t)Bsz * Nn * 2 * Dm * 2);
  half_t* cat1   = (half_t*)alloc((size_t)Bsz * Nn * 2 * Dm * 2);
  half_t* qT     = (half_t*)alloc((size_t)Bsz * Nn * Dm * 2);
  half_t* kT     = (half_t*)alloc((size_t)Bsz * Nn * Dm * 2);
  half_t* vT     = (half_t*)alloc((size_t)Bsz * Nn * Dm * 2);
  half_t* qh     = (half_t*)alloc((size_t)Bsz * Nn * Dm * 2);
  half_t* kh     = (half_t*)alloc((size_t)Bsz * Nn * Dm * 2);
  half_t* vhT    = (half_t*)alloc((size_t)Bsz * Nn * Dm * 2);
  half_t* oT     = (half_t*)alloc((size_t)Bsz * Nn * Dm * 2);
  half_t* h1     = (half_t*)alloc((size_t)Bsz * Nn * 2 * Dm * 2);
  half_t* f0T    = (half_t*)alloc((size_t)Bsz * Nn * Dm * 2);
  half_t* f1T    = (half_t*)alloc((size_t)Bsz * Nn * Dm * 2);
  float*  Smat   = (float*)alloc((size_t)16 * 512 * 512 * 4);
  half_t* Pmat   = (half_t*)alloc((size_t)16 * 512 * 512 * 2);
  float*  Z0     = (float*)alloc((size_t)4 * 513 * 513 * 4);
  float*  uv     = (float*)alloc((size_t)2 * 4 * 513 * 4);
  float*  u      = uv;
  float*  v      = uv + 4 * 513;

  auto cvt = [&](const float* s, half_t* d, long long n) {
    cvt_f32_to_f16<<<dim3((unsigned)((n + 255) / 256)), 256, 0, stream>>>(s, d, n);
  };
  cvt(proj_w,  projW,  (long long)Ll * 3 * Dm * Dm);
  cvt(merge_w, mergeW, (long long)Ll * Dm * Dm);
  cvt(mlp1_w,  mlp1W,  (long long)Ll * 2 * Dm * 2 * Dm);
  cvt(mlp2_w,  mlp2W,  (long long)Ll * Dm * 2 * Dm);
  cvt(final_w, finW,   (long long)Dm * Dm);

  transpose_in<<<2048, 256, 0, stream>>>(x0_in, cat0);
  transpose_in<<<2048, 256, 0, stream>>>(x1_in, cat1);

  auto gemm = [&](const half_t* W, long long wgs, int ldw,
                  const half_t* X, long long xgs, int ldx,
                  const float* bias, half_t* oH, float* oF,
                  int gdiv, long long ogsA, long long ogsB,
                  long long ostr, long long nstr, long long ooff,
                  int O, int Ncols, int G, int K, float alpha, int res) {
    dim3 grid(Ncols / 64, O / 16, G);
    wmma_gemm_kernel<<<grid, 32, 0, stream>>>(W, wgs, ldw, X, xgs, ldx, bias,
                                              oH, oF, gdiv, ogsA, ogsB,
                                              ostr, nstr, ooff, K, alpha, res);
  };

  const long long catGS = (long long)Nn * 2 * Dm;  // 512*512 per batch
  const long long tGS   = (long long)Nn * Dm;      // 512*256 per batch

  for (int l = 0; l < Ll; ++l) {
    // ---- attention for both sides (writes only cat[...,256:512]) ----
    for (int side = 0; side < 2; ++side) {
      half_t* catS   = side ? cat1 : cat0;
      half_t* catSrc = (l & 1) ? (side ? cat0 : cat1) : catS;
      // q/k/v projections (read x channels 0..255 of stride-512 cat)
      for (int j = 0; j < 3; ++j) {
        const half_t* Wp  = projW + (size_t)(l * 3 + j) * Dm * Dm;
        const float*  bp  = proj_b + (size_t)(l * 3 + j) * Dm;
        const half_t* Xp  = (j == 0) ? catS : catSrc;
        half_t* outp = (j == 0) ? qT : (j == 1 ? kT : vT);
        gemm(Wp, 0, Dm, Xp, catGS, 2 * Dm, bp, outp, nullptr,
             1, tGS, 0, /*ostr*/1, /*nstr*/Dm, 0,
             Dm, Nn, Bsz, Dm, 1.0f, 0);
      }
      repack_qkv<<<2048, 256, 0, stream>>>(qT, kT, vT, qh, kh, vhT);
      // S[g][nq][m] = (1/8) q·k
      gemm(qh, (long long)Nn * 64, 64, kh, (long long)Nn * 64, 64, nullptr,
           nullptr, Smat, 1, (long long)512 * 512, 0, 512, 1, 0,
           Nn, Nn, 16, 64, 0.125f, 0);
      softmax_rows<<<2048, 128, 0, stream>>>(Smat, Pmat);
      // O[g][nq][dd] = P·V, stored interleaved into oT[b][nq][dd*4+h]
      gemm(Pmat, (long long)512 * 512, 512, vhT, (long long)64 * 512, 512,
           nullptr, oT, nullptr, /*gdiv*/4, /*ogsA*/tGS, /*ogsB*/1,
           /*ostr*/Dm, /*nstr*/4, 0, Nn, 64, 16, Nn, 1.0f, 0);
      // merge -> msg channels 256..511 of catS
      gemm(mergeW + (size_t)l * Dm * Dm, 0, Dm, oT, tGS, Dm,
           merge_b + (size_t)l * Dm, catS, nullptr,
           1, catGS, 0, 1, 2 * Dm, /*ooff*/Dm, Dm, Nn, Bsz, Dm, 1.0f, 0);
    }
    // ---- MLPs (residual into cat[...,0:256]) ----
    for (int side = 0; side < 2; ++side) {
      half_t* catS = side ? cat1 : cat0;
      gemm(mlp1W + (size_t)l * 2 * Dm * 2 * Dm, 0, 2 * Dm,
           catS, catGS, 2 * Dm, mlp1_b + (size_t)l * 2 * Dm,
           h1, nullptr, 1, catGS, 0, 1, 2 * Dm, 0,
           2 * Dm, Nn, Bsz, 2 * Dm, 1.0f, 0);
      bnrelu<<<4096, 256, 0, stream>>>(h1, bn_g + (size_t)l * 2 * Dm,
                                       bn_b + (size_t)l * 2 * Dm, bn_scale);
      gemm(mlp2W + (size_t)l * Dm * 2 * Dm, 0, 2 * Dm,
           h1, catGS, 2 * Dm, mlp2_b + (size_t)l * Dm,
           catS, nullptr, 1, catGS, 0, 1, 2 * Dm, 0,
           Dm, Nn, Bsz, 2 * Dm, 1.0f, /*residual*/1);
    }
  }

  // final projection
  gemm(finW, 0, Dm, cat0, catGS, 2 * Dm, final_b, f0T, nullptr,
       1, tGS, 0, 1, Dm, 0, Dm, Nn, Bsz, Dm, 1.0f, 0);
  gemm(finW, 0, Dm, cat1, catGS, 2 * Dm, final_b, f1T, nullptr,
       1, tGS, 0, 1, Dm, 0, Dm, Nn, Bsz, Dm, 1.0f, 0);

  // scores[b][n][m] = f0·f1 / 16 into Z0 (row stride 513)
  gemm(f0T, tGS, Dm, f1T, tGS, Dm, nullptr, nullptr, Z0,
       1, (long long)513 * 513, 0, 513, 1, 0, Nn, Nn, Bsz, Dm, 0.0625f, 0);
  fill_bins<<<9, 256, 0, stream>>>(Z0, binscore);

  // Sinkhorn
  zero_f32<<<17, 256, 0, stream>>>(uv, 2LL * 4 * 513);
  for (int it = 0; it < 100; ++it) {
    sink_u<<<513, 128, 0, stream>>>(Z0, v, u);
    sink_v<<<dim3(3, 4), 256, 0, stream>>>(Z0, u, v);
  }
  sink_final<<<4112, 256, 0, stream>>>(Z0, u, v, (float*)d_out);
}
